// DecoderLSTMAttention_40054865002694
// MI455X (gfx1250) — compile-verified, matrix operand order checked by reading, not run on hardware
//
#include <hip/hip_runtime.h>
#include <hip/hip_bf16.h>
#include <math.h>

// ---------------------------------------------------------------------------
// Problem constants (from reference): B=32, S=64, T=64, H=512, V=32000, SOS=1
// ---------------------------------------------------------------------------
#define BB   32
#define SS   64
#define TT   64
#define HH   512
#define VV   32000
#define SOSTOK 1

typedef __attribute__((ext_vector_type(16))) __bf16 v16bf;
typedef __attribute__((ext_vector_type(8)))  float  v8f;

__device__ __forceinline__ float sigmoidf_(float x) {
    return 1.0f / (1.0f + expf(-x));
}

// ---------------------------------------------------------------------------
// Pack Wout [V][H] fp32 -> bf16 in WMMA B-fragment order.
// Layout assumption (16-bit B fragment, wave32, 16x16x32):
//   lane L (<16): N = L,     elems 0..7 -> K = kc*32 + 0..7,  elems 8..15 -> K = kc*32+16..23
//   lane L (>=16): N = L-16, elems 0..7 -> K = kc*32 + 8..15, elems 8..15 -> K = kc*32+24..31
// Packed index: (((vt*16)+kc)*32 + lane)*16 + elem   (vt = V-tile, kc = K-chunk)
// ---------------------------------------------------------------------------
__global__ void __launch_bounds__(256) pack_wout_kernel(
    const float* __restrict__ Wout, __bf16* __restrict__ Bpk)
{
    int idx = blockIdx.x * 256 + threadIdx.x;           // (vt,kc,lane)
    const int total = (VV / 16) * (HH / 32) * 32;       // 2000*16*32
    if (idx >= total) return;
    int lane = idx & 31;
    int kc   = (idx >> 5) & 15;
    int vt   = idx >> 9;
    int n     = lane & 15;
    int khalf = (lane < 16) ? 0 : 8;
    const float* src = Wout + (size_t)(vt * 16 + n) * HH + kc * 32;
    __bf16* dst = Bpk + (size_t)idx * 16;
#pragma unroll
    for (int e = 0; e < 8; ++e) {
        dst[e]     = (__bf16)src[khalf + e];
        dst[8 + e] = (__bf16)src[16 + khalf + e];
    }
}

// ---------------------------------------------------------------------------
// Transpose [rows][cols] -> [cols][rows] (one-time, for coalesced gate GEMV)
// ---------------------------------------------------------------------------
__global__ void __launch_bounds__(256) transpose_kernel(
    const float* __restrict__ W, float* __restrict__ WT, int rows, int cols)
{
    size_t idx = (size_t)blockIdx.x * 256 + threadIdx.x;
    size_t total = (size_t)rows * cols;
    if (idx >= total) return;
    int j = (int)(idx / cols);
    int k = (int)(idx % cols);
    WT[(size_t)k * rows + j] = W[idx];
}

// ---------------------------------------------------------------------------
// tk[b,s,:] = enc[b,s,:] @ Wk^T + bk   (loop-invariant key projection, fp32)
// one block per (b,s) row
// ---------------------------------------------------------------------------
__global__ void __launch_bounds__(256) tk_kernel(
    const float* __restrict__ enc, const float* __restrict__ Wk,
    const float* __restrict__ bk, float* __restrict__ tk)
{
    int bs = blockIdx.x;                        // 0 .. B*S-1
    __shared__ float e[HH];
    const float* erow = enc + (size_t)bs * HH;
    for (int i = threadIdx.x; i < HH; i += 256) e[i] = erow[i];
    __syncthreads();
    for (int h = threadIdx.x; h < HH; h += 256) {
        const float* w = Wk + (size_t)h * HH;
        float acc = bk[h];
#pragma unroll 8
        for (int k = 0; k < HH; ++k) acc = fmaf(e[k], w[k], acc);
        tk[(size_t)bs * HH + h] = acc;
    }
}

// ---------------------------------------------------------------------------
// Per-step attention: tq = h@Wq^T+bq; scores = Ws . tanh(tq + tk) + bs;
// softmax; ctx; x = [emb, ctx]. One block per batch element.
// ---------------------------------------------------------------------------
__global__ void __launch_bounds__(256) attn_step_kernel(
    int t,
    const float* __restrict__ hprev, const float* __restrict__ tk,
    const float* __restrict__ enc,   const int*   __restrict__ target,
    const float* __restrict__ embedding,
    const float* __restrict__ Wq, const float* __restrict__ bq,
    const float* __restrict__ Wsv, const float* __restrict__ bsv,
    float* __restrict__ xbuf, float* __restrict__ attn_out)
{
    int b = blockIdx.x;
    __shared__ float hs[HH], tqs[HH], ss[SS];
    for (int i = threadIdx.x; i < HH; i += 256) hs[i] = hprev[b * HH + i];
    __syncthreads();
    // tq
    for (int h = threadIdx.x; h < HH; h += 256) {
        const float* w = Wq + (size_t)h * HH;
        float acc = bq[h];
#pragma unroll 8
        for (int k = 0; k < HH; ++k) acc = fmaf(hs[k], w[k], acc);
        tqs[h] = acc;
    }
    __syncthreads();
    // scores (threads 0..S-1)
    if (threadIdx.x < SS) {
        int s = threadIdx.x;
        const float* tkrow = tk + ((size_t)b * SS + s) * HH;
        float acc = 0.0f;
#pragma unroll 4
        for (int k = 0; k < HH; ++k) acc = fmaf(Wsv[k], tanhf(tqs[k] + tkrow[k]), acc);
        ss[s] = acc + bsv[0];
    }
    __syncthreads();
    // softmax over S (serial in thread 0; S=64, trivial)
    if (threadIdx.x == 0) {
        float mx = ss[0];
        for (int s = 1; s < SS; ++s) mx = fmaxf(mx, ss[s]);
        float sum = 0.0f;
        for (int s = 0; s < SS; ++s) { float ex = expf(ss[s] - mx); ss[s] = ex; sum += ex; }
        float inv = 1.0f / sum;
        for (int s = 0; s < SS; ++s) ss[s] *= inv;
    }
    __syncthreads();
    // write attention output [B,T,S]
    if (threadIdx.x < SS)
        attn_out[((size_t)b * TT + t) * SS + threadIdx.x] = ss[threadIdx.x];
    // ctx + embedding -> x = [emb, ctx]  in xbuf[b][0..2H)
    int tok = (t == 0) ? SOSTOK : target[b * TT + (t - 1)];
    for (int h = threadIdx.x; h < HH; h += 256) {
        float acc = 0.0f;
#pragma unroll
        for (int s = 0; s < SS; ++s)
            acc = fmaf(ss[s], enc[((size_t)b * SS + s) * HH + h], acc);
        xbuf[(size_t)b * (2 * HH) + HH + h] = acc;
        xbuf[(size_t)b * (2 * HH) + h]      = embedding[(size_t)tok * HH + h];
    }
}

// ---------------------------------------------------------------------------
// Per-step LSTM cell: gates = x@Wih^T + h@Whh^T + biases (coalesced via
// pre-transposed weights), then pointwise. Writes h/c (fp32) and h2 as bf16
// row (b*T + t) of h_all for the big WMMA GEMM.
// grid: 2 blocks per batch element (hi split), 256 threads.
// ---------------------------------------------------------------------------
__global__ void __launch_bounds__(256) lstm_step_kernel(
    int t,
    const float* __restrict__ xbuf, const float* __restrict__ hprev,
    const float* __restrict__ cprev,
    const float* __restrict__ WihT, const float* __restrict__ bih,
    const float* __restrict__ WhhT, const float* __restrict__ bhh,
    float* __restrict__ hnext, float* __restrict__ cnext,
    __bf16* __restrict__ h_all)
{
    int b  = blockIdx.x >> 1;
    int hi = ((blockIdx.x & 1) * 256) + threadIdx.x;    // 0..H-1
    __shared__ float xs[2 * HH], hsh[HH];
    for (int i = threadIdx.x; i < 2 * HH; i += 256) xs[i]  = xbuf[(size_t)b * (2 * HH) + i];
    for (int i = threadIdx.x; i < HH;     i += 256) hsh[i] = hprev[b * HH + i];
    __syncthreads();
    float g[4];
#pragma unroll
    for (int gg = 0; gg < 4; ++gg) {
        int j = gg * HH + hi;
        float acc = bih[j] + bhh[j];
#pragma unroll 4
        for (int k = 0; k < 2 * HH; ++k) acc = fmaf(WihT[(size_t)k * (4 * HH) + j], xs[k], acc);
#pragma unroll 4
        for (int k = 0; k < HH; ++k)     acc = fmaf(WhhT[(size_t)k * (4 * HH) + j], hsh[k], acc);
        g[gg] = acc;
    }
    float iv = sigmoidf_(g[0]);
    float fv = sigmoidf_(g[1]);
    float gv = tanhf(g[2]);
    float ov = sigmoidf_(g[3]);
    float c2 = fv * cprev[b * HH + hi] + iv * gv;
    float h2 = ov * tanhf(c2);
    cnext[b * HH + hi] = c2;
    hnext[b * HH + hi] = h2;
    h_all[((size_t)b * TT + t) * HH + hi] = (__bf16)h2;   // row = b*T + t
}

// ---------------------------------------------------------------------------
// Big WMMA GEMM: logits[2048, 32000] = h_all(bf16) @ Wout_packed(bf16) + bout
// Block = 256 threads (8 waves). Block tile: M=32, N=256.
// Each wave: 2 M-tiles x 2 N-tiles = 4 f32 accumulators; every B fragment is
// reused across the two M-tiles (halves L2 B-traffic vs M=16 tiling).
// A tile (32x512 bf16 = 32 KB) staged in LDS, shared by all 8 waves.
// grid = (2048/32, 32000/256) = (64, 125)
// ---------------------------------------------------------------------------
__global__ void __launch_bounds__(256) gemm_logits_kernel(
    const __bf16* __restrict__ Apk,   // [2048][512] bf16, row-major
    const __bf16* __restrict__ Bpk,   // packed Wout fragments
    const float*  __restrict__ bout,
    float* __restrict__ out)          // logits [2048][32000] fp32
{
    __shared__ __bf16 Atile[32 * HH];               // 32 KB
    const int r0 = blockIdx.x * 32;                 // M-tile base row
    const int ntile0 = blockIdx.y * 16;             // first of 16 N-tiles

    // cooperative A load: 16384 bf16 = 2048 x uint4 (rows r0..r0+31 contiguous)
    {
        const uint4* src = (const uint4*)(Apk + (size_t)r0 * HH);
        uint4* dst = (uint4*)Atile;
#pragma unroll
        for (int i = threadIdx.x; i < 2048; i += 256) dst[i] = src[i];
    }
    __syncthreads();

    const int wave = threadIdx.x >> 5;
    const int lane = threadIdx.x & 31;
    const int m     = lane & 15;
    const int khalf = (lane < 16) ? 0 : 8;
    const int vt0   = ntile0 + wave * 2;            // this wave's two N-tiles

    v8f acc00 = {}; v8f acc01 = {};                 // Mtile0 x {N0,N1}
    v8f acc10 = {}; v8f acc11 = {};                 // Mtile1 x {N0,N1}
#pragma unroll 4
    for (int kc = 0; kc < 16; ++kc) {
        union { v16bf v; uint4 q[2]; } a0, a1, b0, b1;
        const __bf16* ap0 = &Atile[m * HH + kc * 32 + khalf];            // rows 0..15
        const __bf16* ap1 = &Atile[(16 + m) * HH + kc * 32 + khalf];     // rows 16..31
        a0.q[0] = *(const uint4*)(ap0);
        a0.q[1] = *(const uint4*)(ap0 + 16);
        a1.q[0] = *(const uint4*)(ap1);
        a1.q[1] = *(const uint4*)(ap1 + 16);
        const __bf16* bp0 = Bpk + ((size_t)(vt0 * 16 + kc) * 32 + lane) * 16;
        const __bf16* bp1 = Bpk + ((size_t)((vt0 + 1) * 16 + kc) * 32 + lane) * 16;
        b0.q[0] = ((const uint4*)bp0)[0];  b0.q[1] = ((const uint4*)bp0)[1];
        b1.q[0] = ((const uint4*)bp1)[0];  b1.q[1] = ((const uint4*)bp1)[1];
        acc00 = __builtin_amdgcn_wmma_f32_16x16x32_bf16(false, a0.v, false, b0.v,
                                                        (short)0, acc00, false, false);
        acc01 = __builtin_amdgcn_wmma_f32_16x16x32_bf16(false, a0.v, false, b1.v,
                                                        (short)0, acc01, false, false);
        acc10 = __builtin_amdgcn_wmma_f32_16x16x32_bf16(false, a1.v, false, b0.v,
                                                        (short)0, acc10, false, false);
        acc11 = __builtin_amdgcn_wmma_f32_16x16x32_bf16(false, a1.v, false, b1.v,
                                                        (short)0, acc11, false, false);
    }
    // C/D layout: lane<16 -> N=lane, M=r ; lane>=16 -> N=lane-16, M=r+8
    const int nloc = lane & 15;
    const int mofs = (lane < 16) ? 0 : 8;
    const int v0 = vt0 * 16 + nloc;
    const int v1 = (vt0 + 1) * 16 + nloc;
    const float bb0 = bout[v0];
    const float bb1 = bout[v1];
#pragma unroll
    for (int r = 0; r < 8; ++r) {
        size_t rowA = (size_t)(r0 + mofs + r);          // Mtile0
        size_t rowB = (size_t)(r0 + 16 + mofs + r);     // Mtile1
        out[rowA * VV + v0] = acc00[r] + bb0;
        out[rowA * VV + v1] = acc01[r] + bb1;
        out[rowB * VV + v0] = acc10[r] + bb0;
        out[rowB * VV + v1] = acc11[r] + bb1;
    }
}

// ---------------------------------------------------------------------------
// In-place log_softmax over V per row (one block per row).
// ---------------------------------------------------------------------------
__global__ void __launch_bounds__(256) logsoftmax_kernel(float* __restrict__ logits)
{
    size_t row = blockIdx.x;
    float* p = logits + row * VV;
    __shared__ float red[256];
    float mx = -INFINITY;
    for (int i = threadIdx.x; i < VV; i += 256) mx = fmaxf(mx, p[i]);
    red[threadIdx.x] = mx; __syncthreads();
    for (int s = 128; s > 0; s >>= 1) {
        if (threadIdx.x < s) red[threadIdx.x] = fmaxf(red[threadIdx.x], red[threadIdx.x + s]);
        __syncthreads();
    }
    mx = red[0]; __syncthreads();
    float sum = 0.0f;
    for (int i = threadIdx.x; i < VV; i += 256) sum += expf(p[i] - mx);
    red[threadIdx.x] = sum; __syncthreads();
    for (int s = 128; s > 0; s >>= 1) {
        if (threadIdx.x < s) red[threadIdx.x] += red[threadIdx.x + s];
        __syncthreads();
    }
    float lse = mx + logf(red[0]);
    for (int i = threadIdx.x; i < VV; i += 256) p[i] = p[i] - lse;
}

// ---------------------------------------------------------------------------
// Copy final hidden/cell state to outputs.
// ---------------------------------------------------------------------------
__global__ void __launch_bounds__(256) finalize_kernel(
    const float* __restrict__ h, const float* __restrict__ c,
    float* __restrict__ out_h, float* __restrict__ out_c)
{
    int i = blockIdx.x * 256 + threadIdx.x;
    if (i < BB * HH) { out_h[i] = h[i]; out_c[i] = c[i]; }
}

// ---------------------------------------------------------------------------
// Host launch
// ---------------------------------------------------------------------------
extern "C" void kernel_launch(void* const* d_in, const int* in_sizes, int n_in,
                              void* d_out, int out_size, void* d_ws, size_t ws_size,
                              hipStream_t stream)
{
    const float* enc       = (const float*)d_in[0];   // [B,S,H]
    const float* enc_h     = (const float*)d_in[1];   // [1,B,H]
    const float* enc_c     = (const float*)d_in[2];   // [1,B,H]
    const int*   target    = (const int*)  d_in[3];   // [B,T]
    const float* embedding = (const float*)d_in[4];   // [V,H]
    const float* Wq  = (const float*)d_in[5];
    const float* bq  = (const float*)d_in[6];
    const float* Wk  = (const float*)d_in[7];
    const float* bk  = (const float*)d_in[8];
    const float* Wsv = (const float*)d_in[9];
    const float* bsv = (const float*)d_in[10];
    const float* Wih = (const float*)d_in[11];        // [4H,2H]
    const float* bih = (const float*)d_in[12];
    const float* Whh = (const float*)d_in[13];        // [4H,H]
    const float* bhh = (const float*)d_in[14];
    const float* Wout = (const float*)d_in[15];       // [V,H]
    const float* bout = (const float*)d_in[16];

    float* out = (float*)d_out;
    // output layout (flat, return order): log_probs[B,T,V], hT[1,B,H], cT[1,B,H], attn[B,T,S]
    const size_t LOGITS_OFF = 0;
    const size_t HT_OFF     = (size_t)BB * TT * VV;                 // 65,536,000
    const size_t CT_OFF     = HT_OFF + (size_t)BB * HH;
    const size_t ATTN_OFF   = CT_OFF + (size_t)BB * HH;

    // workspace layout (bytes, 256-aligned); total ~52 MB
    char* ws = (char*)d_ws;
    __bf16* wout_pk = (__bf16*)(ws + 0);                            // 32,768,000 B
    float*  tkbuf   = (float*) (ws + 32768000);                     //  4,194,304 B
    __bf16* h_all   = (__bf16*)(ws + 36962304);                     //  2,097,152 B
    float*  xbuf    = (float*) (ws + 39059456);                     //    131,072 B
    float*  hbuf    = (float*) (ws + 39190528);                     //    131,072 B (x2 buffers)
    float*  cbuf    = (float*) (ws + 39321600);                     //    131,072 B (x2 buffers)
    float*  WihT    = (float*) (ws + 39452672);                     //  8,388,608 B
    float*  WhhT    = (float*) (ws + 47841280);                     //  4,194,304 B
    (void)ws_size; (void)in_sizes; (void)n_in; (void)out_size;

    // one-time preprocessing
    pack_wout_kernel<<<(VV / 16 * HH / 32 * 32 + 255) / 256, 256, 0, stream>>>(Wout, wout_pk);
    transpose_kernel<<<(int)(((size_t)4 * HH * 2 * HH + 255) / 256), 256, 0, stream>>>(Wih, WihT, 4 * HH, 2 * HH);
    transpose_kernel<<<(int)(((size_t)4 * HH * HH + 255) / 256), 256, 0, stream>>>(Whh, WhhT, 4 * HH, HH);
    tk_kernel<<<BB * SS, 256, 0, stream>>>(enc, Wk, bk, tkbuf);

    // sequential decoder loop (fp32 recurrence; h2 captured as bf16 rows)
    for (int t = 0; t < TT; ++t) {
        const float* hp = (t == 0) ? enc_h : (hbuf + ((t - 1) & 1) * BB * HH);
        const float* cp = (t == 0) ? enc_c : (cbuf + ((t - 1) & 1) * BB * HH);
        float* hn = hbuf + (t & 1) * BB * HH;
        float* cn = cbuf + (t & 1) * BB * HH;
        attn_step_kernel<<<BB, 256, 0, stream>>>(t, hp, tkbuf, enc, target, embedding,
                                                 Wq, bq, Wsv, bsv, xbuf, out + ATTN_OFF);
        lstm_step_kernel<<<2 * BB, 256, 0, stream>>>(t, xbuf, hp, cp, WihT, bih,
                                                     WhhT, bhh, hn, cn, h_all);
    }

    // batched WMMA bf16 output projection over all (b,t) rows
    dim3 ggrid(BB * TT / 32, VV / 256);     // (64, 125)
    gemm_logits_kernel<<<ggrid, 256, 0, stream>>>(h_all, wout_pk, bout, out + LOGITS_OFF);

    // log_softmax in-place
    logsoftmax_kernel<<<BB * TT, 256, 0, stream>>>(out + LOGITS_OFF);

    // final states
    finalize_kernel<<<(BB * HH + 255) / 256, 256, 0, stream>>>(
        hbuf + ((TT - 1) & 1) * BB * HH, cbuf + ((TT - 1) & 1) * BB * HH,
        out + HT_OFF, out + CT_OFF);
}
